// DotProductAttention_4844723110449
// MI455X (gfx1250) — matrix operand order repaired and checked by reading, not compiled, not run on hardware
//
#include <hip/hip_runtime.h>
#include <hip/hip_bf16.h>
#include <stdint.h>

// Geometry (fixed by the reference): B=2, heads=8, T=8, H=W=16, d=64
#define BB   2
#define NH   8
#define LL   2048          // Lq == Lk
#define DD   64
#define NREL 32            // 31 relative offsets, padded to 32

typedef __attribute__((ext_vector_type(16))) __bf16 v16bf;
typedef __attribute__((ext_vector_type(8)))  float  v8f;

union AFrag { v16bf v; uint4 u[2]; };

// ---------------- prep kernels ----------------

__global__ __launch_bounds__(256) void cvt_q_bf16(const float* __restrict__ q,
                                                  __bf16* __restrict__ dst) {
  int i = blockIdx.x * 256 + threadIdx.x;
  dst[i] = (__bf16)(q[i] * 0.125f);        // fold 1/sqrt(64) into Q
}

__global__ __launch_bounds__(256) void cvt_k_bf16(const float* __restrict__ k,
                                                  __bf16* __restrict__ dst) {
  int i = blockIdx.x * 256 + threadIdx.x;
  dst[i] = (__bf16)k[i];
}

// V [bh, l, d] f32  ->  Vt [bh, d, l] bf16   (so P·V B-fragments load contiguously)
__global__ __launch_bounds__(256) void cvt_vT_bf16(const float* __restrict__ v,
                                                   __bf16* __restrict__ dst) {
  int i = blockIdx.x * 256 + threadIdx.x;   // over BB*NH*DD*LL
  int l  = i & (LL - 1);
  int d  = (i >> 11) & (DD - 1);
  int bh = i >> 17;
  dst[i] = (__bf16)v[((size_t)bh * LL + l) * DD + d];
}

// R[bh, l, m] = sum_d X[bh, l, d] * rel[m, d]   (m in 0..30, slot 31 zero)
__global__ __launch_bounds__(256) void rel_proj(const float* __restrict__ x,
                                                const float* __restrict__ rel,
                                                float* __restrict__ dst) {
  int i = blockIdx.x * 256 + threadIdx.x;   // over BB*NH*LL*NREL
  int m   = i & (NREL - 1);
  int row = i >> 5;
  float acc = 0.f;
  if (m < 31) {
    const float* xr = x + (size_t)row * DD;
    const float* rr = rel + m * DD;
#pragma unroll
    for (int d = 0; d < DD; ++d) acc += xr[d] * rr[d];
  }
  dst[i] = acc;
}

// ---------------- flash attention kernel ----------------
// 8 waves / block share one (b,h); each wave owns a 16-row Q tile.
// K/V chunks are staged once per block into LDS via CDNA5 async-to-LDS copies
// (ASYNCcnt), double-buffered against compute.

// LDS layout (single shared object -> allocated at LDS offset 0):
//   [    0,  8192) : P staging, 1KB per wave
//   [ 8192, 16384) : K tiles, two 4KB buffers (32 keys x 64 bf16)
//   [16384, 24576) : Vt tiles, two 4KB buffers (64 d x 32 keys bf16)
#define SM_PSTAGE 0
#define SM_KBUF   8192
#define SM_VBUF   16384
#define SM_TOTAL  24576

__device__ __forceinline__ void async_b128_to_lds(uint32_t lds_off, const void* gp) {
  asm volatile("global_load_async_to_lds_b128 %0, %1, off"
               :: "v"(lds_off), "v"((unsigned long long)(uintptr_t)gp)
               : "memory");
}

__global__ __launch_bounds__(256) void attn_flash(
    const __bf16* __restrict__ qbf, const __bf16* __restrict__ kbf,
    const __bf16* __restrict__ vtbf,
    const float* __restrict__ Rq, const float* __restrict__ Sq,
    const float* __restrict__ Rk, const float* __restrict__ Sk,
    const float* __restrict__ mask, const float* __restrict__ qorig,
    float* __restrict__ out) {
  __shared__ __align__(16) uint8_t smem[SM_TOTAL];

  const int tid  = threadIdx.x;
  const int lane = tid & 31;
  const int wave = tid >> 5;
  const int g    = blockIdx.x * 8 + wave;   // 0..2047 q-tiles
  const int bh   = g >> 7;                  // uniform across the block
  const int qt   = g & 127;
  const int qbase = qt << 4;
  const int b    = bh >> 3;
  const int half = lane >> 4;
  const int n    = lane & 15;               // key column / d column within 16-group
  const int xq   = (qbase & 255) >> 4;      // query spatial row (constant per tile)

  // --- Q A-fragments (bf16, pre-scaled) ---
  AFrag qa[2];
  {
    const __bf16* qrow = qbf + ((size_t)(bh * LL + qbase + n)) * DD;  // A row = lane%16
#pragma unroll
    for (int c = 0; c < 2; ++c) {
      qa[c].u[0] = *(const uint4*)(qrow + c * 32 + half * 8);
      qa[c].u[1] = *(const uint4*)(qrow + c * 32 + 16 + half * 8);
    }
  }

  // --- Rq gather is chunk-invariant: hoist (index = y_k - y_q + 15 = n - m + 15) ---
  float rqv[8];
#pragma unroll
  for (int j = 0; j < 8; ++j) {
    int m = j + 8 * half;
    rqv[j] = Rq[((size_t)(bh * LL + qbase + m)) * NREL + (n - m + 15)];
  }

  v8f o0 = {}, o1 = {}, o2 = {}, o3 = {};
  float mrow[8], lrow[8];
#pragma unroll
  for (int j = 0; j < 8; ++j) { mrow[j] = -1e30f; lrow[j] = 0.f; }

  __bf16* myp = (__bf16*)(smem + SM_PSTAGE) + wave * 16 * 32;

  // block-cooperative async stage of one 32-key chunk (K: 4KB, Vt: 4KB)
  auto stage = [&](int kb, int p) {
    {  // K tile: 32 rows x 64 bf16, thread copies 16B
      int r  = tid >> 3;
      int co = (tid & 7) * 8;
      const __bf16* gp = kbf + ((size_t)(bh * LL + kb + r)) * DD + co;
      async_b128_to_lds((uint32_t)(SM_KBUF + p * 4096 + tid * 16), gp);
    }
    {  // Vt tile: 64 rows(d) x 32 keys bf16, thread copies 16B
      int d  = tid >> 2;
      int ko = (tid & 3) * 8;
      const __bf16* gp = vtbf + ((size_t)(bh * DD + d)) * LL + kb + ko;
      async_b128_to_lds((uint32_t)(SM_VBUF + p * 4096 + tid * 16), gp);
    }
  };

  stage(0, 0);
  int par = 0;

  for (int kb = 0; kb < LL; kb += 32) {
    // my async slices of chunk kb have landed; barrier makes all slices visible
    asm volatile("s_wait_asynccnt 0" ::: "memory");
    __syncthreads();
    if (kb + 32 < LL) stage(kb + 32, par ^ 1);   // overlap next copy with compute

    const __bf16* kt = (const __bf16*)(smem + SM_KBUF + par * 4096);
    const __bf16* vt = (const __bf16*)(smem + SM_VBUF + par * 4096);

    // ---- S = Q K^T (two 16x16 tiles over 32 keys, contraction d=64) ----
    v8f s0 = {}, s1 = {};
#pragma unroll
    for (int c = 0; c < 2; ++c) {
      AFrag kf0, kf1;
      const __bf16* kr0 = kt + (size_t)(n)      * DD + c * 32 + half * 8;
      const __bf16* kr1 = kt + (size_t)(16 + n) * DD + c * 32 + half * 8;
      kf0.u[0] = *(const uint4*)(kr0);
      kf0.u[1] = *(const uint4*)(kr0 + 16);
      kf1.u[0] = *(const uint4*)(kr1);
      kf1.u[1] = *(const uint4*)(kr1 + 16);
      s0 = __builtin_amdgcn_wmma_f32_16x16x32_bf16(false, qa[c].v, false, kf0.v,
                                                   (short)0, s0, false, false);
      s1 = __builtin_amdgcn_wmma_f32_16x16x32_bf16(false, qa[c].v, false, kf1.v,
                                                   (short)0, s1, false, false);
    }

    // ---- bias + mask (closed-form skew: abs[i,j] = rel[i, j-i+15]) ----
    const int xk0 = (kb & 255) >> 4;
    float sv[2][8];
#pragma unroll
    for (int s = 0; s < 2; ++s) {
      const int xk  = xk0 + s;
      const int key = kb + s * 16 + n;                 // y_k == n
#pragma unroll
      for (int j = 0; j < 8; ++j) {
        const int m = j + 8 * half;                    // y_q == m
        const int q = qbase + m;
        float bias = rqv[j]
                   + Sq[((size_t)(bh * LL + q)) * NREL + (xk - xq + 15)]
                   + Rk[((size_t)(bh * LL + key)) * NREL + (m - n + 15)]
                   + Sk[((size_t)(bh * LL + key)) * NREL + (xq - xk + 15)];
        float mv = mask[(size_t)b * LL * LL + (size_t)q * LL + key];
        float acc = (s == 0) ? s0[j] : s1[j];
        sv[s][j] = acc + bias - 1e9f * (1.0f - mv);
      }
    }

    // ---- online softmax (row stats across the 16-lane half-groups) ----
#pragma unroll
    for (int j = 0; j < 8; ++j) {
      float t = fmaxf(sv[0][j], sv[1][j]);
      t = fmaxf(t, __shfl_xor(t, 1, 32));
      t = fmaxf(t, __shfl_xor(t, 2, 32));
      t = fmaxf(t, __shfl_xor(t, 4, 32));
      t = fmaxf(t, __shfl_xor(t, 8, 32));
      float mn    = fmaxf(mrow[j], t);
      float alpha = __expf(mrow[j] - mn);
      mrow[j] = mn;
      float p0 = __expf(sv[0][j] - mn);
      float p1 = __expf(sv[1][j] - mn);
      float rs = p0 + p1;
      rs += __shfl_xor(rs, 1, 32);
      rs += __shfl_xor(rs, 2, 32);
      rs += __shfl_xor(rs, 4, 32);
      rs += __shfl_xor(rs, 8, 32);
      lrow[j] = lrow[j] * alpha + rs;
      o0[j] *= alpha; o1[j] *= alpha; o2[j] *= alpha; o3[j] *= alpha;
      const int m = j + 8 * half;
      myp[m * 32 + n]      = (__bf16)p0;               // C-layout -> row-major LDS
      myp[m * 32 + 16 + n] = (__bf16)p1;
    }

    // wave-private LDS round trip; DS ops are in-order per wave
    asm volatile("s_wait_dscnt 0" ::: "memory");

    AFrag pa;                                          // P as 16x32 bf16 A-fragment
    pa.u[0] = *(const uint4*)(myp + n * 32 + half * 8);
    pa.u[1] = *(const uint4*)(myp + n * 32 + 16 + half * 8);

    // ---- O += P V  (4 d-chunks of 16, contraction = 32 keys) ----
#pragma unroll
    for (int dc = 0; dc < 4; ++dc) {
      AFrag vf;
      const __bf16* vr = vt + (size_t)(dc * 16 + n) * 32 + half * 8;
      vf.u[0] = *(const uint4*)(vr);
      vf.u[1] = *(const uint4*)(vr + 16);
      v8f& od = (dc == 0) ? o0 : (dc == 1) ? o1 : (dc == 2) ? o2 : o3;
      od = __builtin_amdgcn_wmma_f32_16x16x32_bf16(false, pa.v, false, vf.v,
                                                   (short)0, od, false, false);
    }

    // all waves done reading this buffer before it is overwritten two chunks later
    __syncthreads();
    par ^= 1;
  }

  // ---- epilogue: normalize, residual add, store f32 ----
#pragma unroll
  for (int dc = 0; dc < 4; ++dc) {
    const v8f& od = (dc == 0) ? o0 : (dc == 1) ? o1 : (dc == 2) ? o2 : o3;
#pragma unroll
    for (int j = 0; j < 8; ++j) {
      const int m = j + 8 * half;
      size_t idx = ((size_t)(bh * LL + qbase + m)) * DD + dc * 16 + n;
      out[idx] = od[j] / lrow[j] + qorig[idx];
    }
  }
}

// ---------------- host launcher ----------------

extern "C" void kernel_launch(void* const* d_in, const int* in_sizes, int n_in,
                              void* d_out, int out_size, void* d_ws, size_t ws_size,
                              hipStream_t stream) {
  const float* q    = (const float*)d_in[0];
  const float* k    = (const float*)d_in[1];
  const float* v    = (const float*)d_in[2];
  const float* mask = (const float*)d_in[3];
  const float* rhq  = (const float*)d_in[4];
  const float* rwq  = (const float*)d_in[5];
  const float* rhk  = (const float*)d_in[6];
  const float* rwk  = (const float*)d_in[7];
  float* out = (float*)d_out;

  char* ws = (char*)d_ws;
  const size_t NE = (size_t)BB * NH * LL * DD;        // 2M elements
  const size_t NR = (size_t)BB * NH * LL * NREL;      // 1M elements

  __bf16* qbf  = (__bf16*)ws;                   // 4 MB
  __bf16* kbf  = (__bf16*)(ws + NE * 2);        // 4 MB
  __bf16* vtbf = (__bf16*)(ws + NE * 4);        // 4 MB
  float*  Rq   = (float*)(ws + NE * 6);         // 4 MB
  float*  Sq   = Rq + NR;
  float*  Rk   = Sq + NR;
  float*  Sk   = Rk + NR;                       // total 28 MB

  // bf16 operand prep
  cvt_q_bf16 <<<NE / 256, 256, 0, stream>>>(q, qbf);
  cvt_k_bf16 <<<NE / 256, 256, 0, stream>>>(k, kbf);
  cvt_vT_bf16<<<NE / 256, 256, 0, stream>>>(v, vtbf);

  // relative-position projection tables (closed-form skew bias)
  rel_proj<<<NR / 256, 256, 0, stream>>>(q, rwq, Rq);
  rel_proj<<<NR / 256, 256, 0, stream>>>(q, rhq, Sq);
  rel_proj<<<NR / 256, 256, 0, stream>>>(k, rwk, Rk);
  rel_proj<<<NR / 256, 256, 0, stream>>>(k, rhk, Sk);

  // flash attention: 2048 q-tiles of 16 rows, 8 waves / block
  attn_flash<<<(BB * NH * (LL / 16)) / 8, 256, 0, stream>>>(
      qbf, kbf, vtbf, Rq, Sq, Rk, Sk, mask, q, out);
}
